// ChildSumTreeLSTMCell_30855045054488
// MI455X (gfx1250) — matrix operand order repaired and checked by reading, not compiled, not run on hardware
//
#include <hip/hip_runtime.h>

#define N_NODES 100000
#define M_EDGES 99999
#define ENC 512
#define KIOU 1024
#define NOUT 1536

typedef __attribute__((ext_vector_type(16))) __bf16   v16bf;
typedef __attribute__((ext_vector_type(8)))  float    v8f;
typedef __attribute__((ext_vector_type(8)))  unsigned int v8u;

// ---------- helpers ----------
__device__ __forceinline__ unsigned short f2bf(float f) {
    unsigned int u = __float_as_uint(f);
    u += 0x7FFFu + ((u >> 16) & 1u);          // round-to-nearest-even
    return (unsigned short)(u >> 16);
}
__device__ __forceinline__ float bf2f(unsigned short s) {
    return __uint_as_float(((unsigned int)s) << 16);
}
__device__ __forceinline__ float sigmoidf(float v) { return 1.f / (1.f + expf(-v)); }

__device__ __forceinline__ v8f wmma_bf16(v8u a, v8u b, v8f c) {
    return __builtin_amdgcn_wmma_f32_16x16x32_bf16(
        false, __builtin_bit_cast(v16bf, a),
        false, __builtin_bit_cast(v16bf, b),
        (short)0, c, false, false);
}

// A-operand fragment from an LDS bf16 plane (rows x Kstride ushorts).
// A layout used here (contiguous k, same as B packing): lanes 0-15: M=lane, K=k0..k0+15;
// lanes 16-31: M=lane-16, K=k0+16..k0+31; pairs packed per VGPR.
__device__ __forceinline__ v8u load_a_frag(const unsigned short* plane, int kstride, int k0) {
    const int lane = threadIdx.x & 31;
    const int half = lane >> 4;
    const int m    = lane & 15;
    const unsigned int* rp = (const unsigned int*)(plane + (size_t)m * kstride);
    const int p = (k0 + half * 16) >> 1;
    v8u a;
    a[0] = rp[p + 0]; a[1] = rp[p + 1]; a[2] = rp[p + 2]; a[3] = rp[p + 3];
    a[4] = rp[p + 4]; a[5] = rp[p + 5]; a[6] = rp[p + 6]; a[7] = rp[p + 7];
    return a;
}

// 3-term split-precision MMA: (Ahi+Alo)*(Bhi+Blo) ~= AhiBhi + AloBhi + AhiBlo
__device__ __forceinline__ void mma3(v8f& acc, v8u ahi, v8u alo,
                                     const unsigned int* __restrict__ bhi,
                                     const unsigned int* __restrict__ blo, size_t fo) {
    v8u bh = *(const v8u*)(bhi + fo);
    v8u bl = *(const v8u*)(blo + fo);
    acc = wmma_bf16(ahi, bh, acc);
    acc = wmma_bf16(alo, bh, acc);
    acc = wmma_bf16(ahi, bl, acc);
}

// ---------- weight pre-swizzle (once per launch; ~7 MB, L2 resident) ----------
// B fragment storage: ((kt*NT + nt)*32 + lane)*8 u32; lane 0-15: N=lane, K=kt*32+2v(+1);
// lanes 16-31: N=lane-16, K=kt*32+16+2v(+1).
__global__ void __launch_bounds__(256)
convert_uf(const float* __restrict__ w, unsigned int* __restrict__ hi,
           unsigned int* __restrict__ lo) {
    int t = blockIdx.x * 256 + threadIdx.x;          // 16 kt * 32 nt * 32 lanes
    if (t >= 16 * 32 * 32) return;
    int lane = t & 31;
    int nt   = (t >> 5) & 31;
    int kt   = t >> 10;
    int n    = nt * 16 + (lane & 15);
    int kb   = kt * 32 + (lane >> 4) * 16;
    size_t o = (size_t)t * 8;
#pragma unroll
    for (int v = 0; v < 8; ++v) {
        int k = kb + 2 * v;
        float f0 = w[(size_t)n * ENC + k];
        float f1 = w[(size_t)n * ENC + k + 1];
        unsigned short h0 = f2bf(f0), h1 = f2bf(f1);
        unsigned short l0 = f2bf(f0 - bf2f(h0)), l1 = f2bf(f1 - bf2f(h1));
        hi[o + v] = (unsigned)h0 | ((unsigned)h1 << 16);
        lo[o + v] = (unsigned)l0 | ((unsigned)l1 << 16);
    }
}

__global__ void __launch_bounds__(256)
convert_wu(const float* __restrict__ W, const float* __restrict__ U,
           unsigned int* __restrict__ hi, unsigned int* __restrict__ lo) {
    int t = blockIdx.x * 256 + threadIdx.x;          // 32 kt * 96 nt * 32 lanes
    if (t >= 32 * 96 * 32) return;
    int lane = t & 31;
    int nt   = (t >> 5) % 96;
    int kt   = (t >> 5) / 96;
    int n    = nt * 16 + (lane & 15);
    int kb   = kt * 32 + (lane >> 4) * 16;           // whole fragment is on one side of 512
    const float* src = (kb < ENC) ? (W + (size_t)n * ENC) : (U + (size_t)n * ENC - ENC);
    size_t o = (size_t)t * 8;
#pragma unroll
    for (int v = 0; v < 8; ++v) {
        int k = kb + 2 * v;
        float f0 = src[k];
        float f1 = src[k + 1];
        unsigned short h0 = f2bf(f0), h1 = f2bf(f1);
        unsigned short l0 = f2bf(f0 - bf2f(h0)), l1 = f2bf(f1 - bf2f(h1));
        hi[o + v] = (unsigned)h0 | ((unsigned)h1 << 16);
        lo[o + v] = (unsigned)l0 | ((unsigned)l1 << 16);
    }
}

// ---------- edge kernel: f = sigma(h[child]@Uf^T + b); scatter sums ----------
__global__ void __launch_bounds__(256)
edge_kernel(const float* __restrict__ h, const float* __restrict__ c,
            const int* __restrict__ child, const int* __restrict__ parent,
            const unsigned int* __restrict__ bhi, const unsigned int* __restrict__ blo,
            const float* __restrict__ ufb,
            float* __restrict__ h_out /* h_tilde accum */,
            float* __restrict__ c_out /* c_red accum  */,
            int* __restrict__ deg) {
    __shared__ unsigned short Ahi[16][ENC];
    __shared__ unsigned short Alo[16][ENC];
    __shared__ int sChild[16], sParent[16];
    const int tid = threadIdx.x;
    const int e0  = blockIdx.x * 16;

    if (tid < 16) {
        int e = e0 + tid, ch = 0, pa = 0;
        if (e < M_EDGES) { ch = child[e]; pa = parent[e]; }
        sChild[tid] = ch; sParent[tid] = pa;
        if (e < M_EDGES) atomicAdd(&deg[pa], 1);
    }
    __syncthreads();

    // gather h[child] rows, scatter into h_tilde, build bf16 hi/lo planes
    for (int idx = tid; idx < 16 * ENC; idx += 256) {
        int m = idx >> 9, k = idx & (ENC - 1);
        int e = e0 + m;
        float v = 0.f;
        if (e < M_EDGES) {
            v = h[(size_t)sChild[m] * ENC + k];
            atomicAdd(&h_out[(size_t)sParent[m] * ENC + k], v);
        }
        unsigned short hv = f2bf(v);
        Ahi[m][k] = hv;
        Alo[m][k] = f2bf(v - bf2f(hv));
    }
    __syncthreads();

    const int lane = tid & 31, wave = tid >> 5;
    const int half = lane >> 4, ln = lane & 15;
    v8f acc[4] = {};
    for (int kt = 0; kt < ENC / 32; ++kt) {
        v8u ahi = load_a_frag(&Ahi[0][0], ENC, kt * 32);
        v8u alo = load_a_frag(&Alo[0][0], ENC, kt * 32);
#pragma unroll
        for (int g = 0; g < 4; ++g) {
            int nt = wave * 4 + g;
            size_t fo = (((size_t)kt * 32 + nt) * 32 + lane) * 8;
            mma3(acc[g], ahi, alo, bhi, blo, fo);
        }
    }

    // f = sigmoid(acc + b); c_red[parent] += f * c[child]
#pragma unroll
    for (int g = 0; g < 4; ++g) {
        int col = (wave * 4 + g) * 16 + ln;
        float bias = ufb[col];
#pragma unroll
        for (int v = 0; v < 8; ++v) {
            int m = v + half * 8;
            int e = e0 + m;
            if (e < M_EDGES) {
                float f  = sigmoidf(acc[g][v] + bias);
                float cs = c[(size_t)sChild[m] * ENC + col];
                atomicAdd(&c_out[(size_t)sParent[m] * ENC + col], f * cs);
            }
        }
    }
}

// ---------- node kernel: fused iou GEMM (K=1024) + LSTM epilogue ----------
__global__ void __launch_bounds__(256)
node_kernel(const float* __restrict__ x,
            float* __restrict__ h_out /* in: h_tilde, out: h_new */,
            float* __restrict__ c_out /* in: c_red,  out: c_new */,
            const int* __restrict__ deg,
            const unsigned int* __restrict__ bhi, const unsigned int* __restrict__ blo,
            const float* __restrict__ b_iou) {
    __shared__ unsigned short Ahi[16][KIOU];
    __shared__ unsigned short Alo[16][KIOU];
    const int tid = threadIdx.x;
    const int n0  = blockIdx.x * 16;

    // A = [has_child ? 0 : x | has_child ? h_tilde : 0], bf16 hi/lo
    for (int idx = tid; idx < 16 * ENC; idx += 256) {
        int m = idx >> 9, k = idx & (ENC - 1);
        int node = n0 + m;
        bool hc = deg[node] > 0;
        float v1 = hc ? 0.f : x[(size_t)node * ENC + k];
        float v2 = hc ? h_out[(size_t)node * ENC + k] : 0.f;
        unsigned short h1 = f2bf(v1), h2 = f2bf(v2);
        Ahi[m][k]       = h1;  Alo[m][k]       = f2bf(v1 - bf2f(h1));
        Ahi[m][ENC + k] = h2;  Alo[m][ENC + k] = f2bf(v2 - bf2f(h2));
    }
    __syncthreads();

    const int lane = tid & 31, wave = tid >> 5;
    const int half = lane >> 4, ln = lane & 15;
    // wave owns 4 column groups; per group the i/o/u ntiles (cg, cg+32, cg+64)
    v8f ai[4] = {}, ao[4] = {}, au[4] = {};
    for (int kt = 0; kt < KIOU / 32; ++kt) {
        v8u ahi = load_a_frag(&Ahi[0][0], KIOU, kt * 32);
        v8u alo = load_a_frag(&Alo[0][0], KIOU, kt * 32);
#pragma unroll
        for (int g = 0; g < 4; ++g) {
            int cg = wave * 4 + g;
            size_t base = ((size_t)kt * 96) * 32 * 8;
            mma3(ai[g], ahi, alo, bhi, blo, base + ((size_t)(cg)      * 32 + lane) * 8);
            mma3(ao[g], ahi, alo, bhi, blo, base + ((size_t)(cg + 32) * 32 + lane) * 8);
            mma3(au[g], ahi, alo, bhi, blo, base + ((size_t)(cg + 64) * 32 + lane) * 8);
        }
    }

    // in-register LSTM epilogue: c_new = sig(i)*tanh(u) + c_red ; h_new = sig(o)*tanh(c_new)
#pragma unroll
    for (int g = 0; g < 4; ++g) {
        int col = (wave * 4 + g) * 16 + ln;
        float bi = b_iou[col], bo = b_iou[col + ENC], bu = b_iou[col + 2 * ENC];
#pragma unroll
        for (int v = 0; v < 8; ++v) {
            int node = n0 + v + half * 8;
            size_t off = (size_t)node * ENC + col;
            float iv = ai[g][v] + bi;
            float ov = ao[g][v] + bo;
            float uv = au[g][v] + bu;
            float cn = sigmoidf(iv) * tanhf(uv) + c_out[off];
            c_out[off] = cn;
            h_out[off] = sigmoidf(ov) * tanhf(cn);
        }
    }
}

// ---------- workspace layout ----------
static constexpr size_t WS_DEG  = 0;                          // N ints
static constexpr size_t WS_UFHI = 512 * 1024;                 // 16*32*32*8 u32
static constexpr size_t WS_UFLO = WS_UFHI + 524288;
static constexpr size_t WS_WUHI = WS_UFLO + 524288;           // 32*96*32*8 u32
static constexpr size_t WS_WULO = WS_WUHI + 3145728;

extern "C" void kernel_launch(void* const* d_in, const int* in_sizes, int n_in,
                              void* d_out, int out_size, void* d_ws, size_t ws_size,
                              hipStream_t stream) {
    const float* x      = (const float*)d_in[0];
    const float* h      = (const float*)d_in[1];
    const float* c      = (const float*)d_in[2];
    const int*   child  = (const int*)d_in[3];
    const int*   parent = (const int*)d_in[4];
    const float* W_iou  = (const float*)d_in[5];
    const float* U_iou  = (const float*)d_in[6];
    const float* b_iou  = (const float*)d_in[7];
    const float* U_f_w  = (const float*)d_in[8];
    const float* U_f_b  = (const float*)d_in[9];

    float* h_out = (float*)d_out;                 // h_new (h_tilde staged here)
    float* c_out = h_out + (size_t)N_NODES * ENC; // c_new (c_red accumulated here)

    char* ws = (char*)d_ws;
    int*          deg  = (int*)(ws + WS_DEG);
    unsigned int* ufhi = (unsigned int*)(ws + WS_UFHI);
    unsigned int* uflo = (unsigned int*)(ws + WS_UFLO);
    unsigned int* wuhi = (unsigned int*)(ws + WS_WUHI);
    unsigned int* wulo = (unsigned int*)(ws + WS_WULO);

    hipMemsetAsync(d_out, 0, (size_t)out_size * sizeof(float), stream);
    hipMemsetAsync(deg, 0, (size_t)N_NODES * sizeof(int), stream);

    convert_uf<<<64, 256, 0, stream>>>(U_f_w, ufhi, uflo);
    convert_wu<<<384, 256, 0, stream>>>(W_iou, U_iou, wuhi, wulo);

    edge_kernel<<<(M_EDGES + 15) / 16, 256, 0, stream>>>(
        h, c, child, parent, ufhi, uflo, U_f_b, h_out, c_out, deg);

    node_kernel<<<N_NODES / 16, 256, 0, stream>>>(
        x, h_out, c_out, deg, wuhi, wulo, b_iou);
}
